// OpenLSTM_19464791785857
// MI455X (gfx1250) — matrix operand order repaired and checked by reading, not compiled, not run on hardware
//
#include <hip/hip_runtime.h>

// ---------------------------------------------------------------------------
// OpenLSTM for MI455X (gfx1250): latency-bound 512-step recurrence.
//   * f16 data / f32-accum WMMA (v_wmma_f32_16x16x32_f16), wave32 layouts.
//   * 16 persistent WGs; each owns 128 gate columns (32 hidden x 4 gates).
//     Both weight slices LDS-resident (260KB of CDNA5's 320KB/WGP): zero
//     weight traffic inside the sequential loop.
//   * Wave decomposition: wave = 16 batch rows x (4 gates x 16 hidden cols).
//     A fragments (global/L2) AND all 8 B fragments (LDS) double-buffered one
//     K-step ahead -> WMMAs never wait on same-iteration loads.
//   * Accumulators start at 0 (inline SRC2=0 on first WMMA, no splat movs ->
//     no WMMA->VALU hazard NOPs); bias folded into the elementwise cell.
//   * h double-buffered in global (L2-resident, 64KB f16); one agent-scope
//     atomic barrier per step (fence + syncthreads + release/acquire ctr).
// ---------------------------------------------------------------------------

typedef _Float16 v16h __attribute__((ext_vector_type(16)));
typedef _Float16 v8h  __attribute__((ext_vector_type(8)));
typedef float    v8f  __attribute__((ext_vector_type(8)));

constexpr int S = 512, B = 64, F = 512, H = 512;
constexpr int NWG  = 16;       // workgroups in recurrent kernel
constexpr int HC   = 32;       // hidden columns per WG
constexpr int COLS = 4 * HC;   // gate columns per WG (i,f,g,o slices)
constexpr int LDW  = F + 8;    // padded LDS weight row stride (f16 elems)

__device__ __forceinline__ float sigm(float x) {
  return 1.0f / (1.0f + __expf(-x));
}
__device__ __forceinline__ float tanh_fast(float x) {
  return 2.0f / (1.0f + __expf(-2.0f * x)) - 1.0f;
}

// A fragment (16x32 f16, M x K): lane holds row m = lane&15;
// elems 0..15 = K kb..kb+15, kb = (lane>>4)*16.
__device__ __forceinline__ v16h load_a(const _Float16* __restrict__ p) {
  const v8h* p0 = (const v8h*)p;
  const v8h* p1 = (const v8h*)(p + 8);
  v8h lo = *p0, hi = *p1;
  return __builtin_shufflevector(lo, hi, 0, 1, 2, 3, 4, 5, 6, 7,
                                 8, 9, 10, 11, 12, 13, 14, 15);
}

// B fragment (32x16 f16, K x N) from LDS slice stored [col][k]:
// lane holds col n = lane&15; 16 contiguous K at kb = (lane>>4)*16.
__device__ __forceinline__ v16h load_b(const _Float16* col_base, int kb) {
  const v8h* p = (const v8h*)(col_base + kb);
  v8h lo = p[0], hi = p[1];
  return __builtin_shufflevector(lo, hi, 0, 1, 2, 3, 4, 5, 6, 7,
                                 8, 9, 10, 11, 12, 13, 14, 15);
}

__device__ __forceinline__ v8f wmma_f16(v16h a, v16h b, v8f c) {
  return __builtin_amdgcn_wmma_f32_16x16x32_f16(false, a, false, b,
                                                (short)0, c, false, false);
}

__global__ void lstm_init(const float* __restrict__ seq,
                          _Float16* __restrict__ seq16,
                          _Float16* __restrict__ hbuf,
                          unsigned* __restrict__ ctr) {
  const size_t stride = (size_t)gridDim.x * blockDim.x;
  const size_t tid0 = (size_t)blockIdx.x * blockDim.x + threadIdx.x;
  const size_t n = (size_t)S * B * F;
  for (size_t i = tid0; i < n; i += stride)
    seq16[i] = (_Float16)seq[i];
  for (size_t i = tid0; i < (size_t)2 * B * H; i += stride)
    hbuf[i] = (_Float16)0.0f;
  if (tid0 == 0) *ctr = 0u;
}

__global__ __launch_bounds__(256, 1)
void lstm_recurrent(const _Float16* __restrict__ seq16,
                    const float* __restrict__ w_ih,
                    const float* __restrict__ w_hh,
                    const float* __restrict__ b_ih,
                    const float* __restrict__ b_hh,
                    float* __restrict__ hs_out,
                    float* __restrict__ cs_out,
                    _Float16* __restrict__ hbuf,   // 2 * B*H (double buffer)
                    unsigned* __restrict__ ctr) {
  extern __shared__ char smem[];
  _Float16* sWih = (_Float16*)smem;              // [COLS][LDW]
  _Float16* sWhh = sWih + COLS * LDW;            // [COLS][LDW]

  const int wg   = blockIdx.x;          // 0..15: owns hidden cols wg*32..+32
  const int tid  = threadIdx.x;         // 256 threads = 8 waves
  const int lane = tid & 31;
  const int wv   = tid >> 5;

  // ---- one-time: stage this WG's weight slices into LDS as f16 ----
  // local gate col c in [0,128): gate = c>>5, hidden j = c&31,
  // global weight row = gate*H + wg*HC + j   (PyTorch gate order i,f,g,o)
  for (int idx = tid; idx < COLS * F; idx += 256) {
    int c = idx >> 9;            // F == 512
    int k = idx & (F - 1);
    int gr = ((c >> 5) * H) + wg * HC + (c & 31);
    sWih[c * LDW + k] = (_Float16)w_ih[(size_t)gr * F + k];
    sWhh[c * LDW + k] = (_Float16)w_hh[(size_t)gr * H + k];
  }
  __syncthreads();

  // Wave tile: rows rb..rb+16 (batch), hidden cols jbase..jbase+16, 4 gates.
  const int rb    = (wv >> 1) * 16;      // 4 row groups
  const int jbase = (wv & 1) * 16;       // 2 hidden-col groups
  const int aM    = lane & 15;           // A-frag row within tile
  const int akb   = (lane >> 4) << 4;    // A-frag K base (0 / 16)
  const int bkb   = (lane >> 4) << 4;    // B-frag K base (0 / 16)
  const int nl    = jbase + (lane & 15); // lane's hidden col within WG slice
  const int hi8   = (lane >> 4) << 3;    // D-frag row offset (0 / 8)

  // Per-gate LDS column bases and folded bias for this lane.
  const _Float16* bih_base[4];
  const _Float16* bhh_base[4];
  float biasv[4];
#pragma unroll
  for (int g = 0; g < 4; ++g) {
    int c = g * HC + nl;                       // local gate col
    bih_base[g] = sWih + (size_t)c * LDW;
    bhh_base[g] = sWhh + (size_t)c * LDW;
    int gr = g * H + wg * HC + nl;             // global gate row
    biasv[g] = b_ih[gr] + b_hh[gr];
  }

  // c-state registers: lane owns rows rb+hi8+0..7 at hidden col wg*HC+nl.
  float cst[8];
#pragma unroll
  for (int r = 0; r < 8; ++r) cst[r] = 0.0f;

  const int hcol = wg * HC + nl;

  for (int s = 0; s < S; ++s) {
    const _Float16* x  = seq16 + (size_t)s * (B * F);
    const _Float16* hp = hbuf + (size_t)(s & 1) * (B * H);
    _Float16*       hn = hbuf + (size_t)((s + 1) & 1) * (B * H);

    const _Float16* xrow = x  + (size_t)(rb + aM) * F + akb;
    const _Float16* hrow = hp + (size_t)(rb + aM) * H + akb;

    // Both chains start at 0 -> first WMMA of each chain gets inline SRC2=0
    // (no accumulator splat movs, no WMMA->VALU hazard NOPs). Bias is added
    // in the elementwise phase below.
    v8f accx[4], acch[4];
#pragma unroll
    for (int g = 0; g < 4; ++g) {
#pragma unroll
      for (int i = 0; i < 8; ++i) { accx[g][i] = 0.0f; acch[g][i] = 0.0f; }
    }

    // Fully software-pipelined K loop: A frags (global) and all 8 B frags
    // (LDS) for kt+32 are issued before the 8 WMMAs that consume the kt
    // fragments. In the fully unrolled loop the cur=next copies become
    // register rotation, not movs.
    v16h axc = load_a(xrow);
    v16h ahc = load_a(hrow);
    v16h bihc[4], bhhc[4];
#pragma unroll
    for (int g = 0; g < 4; ++g) {
      bihc[g] = load_b(bih_base[g], bkb);
      bhhc[g] = load_b(bhh_base[g], bkb);
    }

#pragma unroll
    for (int kt = 0; kt < F; kt += 32) {
      v16h axn, ahn, bihn[4], bhhn[4];
      if (kt + 32 < F) {                 // compile-time after full unroll
        axn = load_a(xrow + kt + 32);
        ahn = load_a(hrow + kt + 32);
#pragma unroll
        for (int g = 0; g < 4; ++g) {
          bihn[g] = load_b(bih_base[g] + kt + 32, bkb);
          bhhn[g] = load_b(bhh_base[g] + kt + 32, bkb);
        }
      }
#pragma unroll
      for (int g = 0; g < 4; ++g) {
        accx[g] = wmma_f16(axc, bihc[g], accx[g]);
        acch[g] = wmma_f16(ahc, bhhc[g], acch[g]);
      }
      if (kt + 32 < F) {
        axc = axn;
        ahc = ahn;
#pragma unroll
        for (int g = 0; g < 4; ++g) { bihc[g] = bihn[g]; bhhc[g] = bhhn[g]; }
      }
    }

    // ---- LSTM cell, entirely in registers ----
    // D layout: VGPR r -> row m = rb + hi8 + r, col hcol.
#pragma unroll
    for (int r = 0; r < 8; ++r) {
      float gi = accx[0][r] + acch[0][r] + biasv[0];
      float gf = accx[1][r] + acch[1][r] + biasv[1];
      float gg = accx[2][r] + acch[2][r] + biasv[2];
      float go = accx[3][r] + acch[3][r] + biasv[3];
      float c = sigm(gf) * cst[r] + sigm(gi) * tanh_fast(gg);
      cst[r] = c;
      float h = sigm(go) * tanh_fast(c);
      int b = rb + hi8 + r;
      hn[(size_t)b * H + hcol] = (_Float16)h;
      size_t oidx = (size_t)s * (B * H) + (size_t)b * H + hcol;
      hs_out[oidx] = h;
      cs_out[oidx] = c;
    }

    // prefetch next step's x while we wait at the barrier
    if (s + 1 < S) {
      const char* nx = (const char*)(seq16 + (size_t)(s + 1) * (B * F));
      __builtin_prefetch(nx + (size_t)tid * 256, 0, 0);
    }

    // ---- per-step grid barrier (generation counter, agent scope) ----
    __threadfence();   // make this thread's h stores agent-visible
    __syncthreads();
    if (tid == 0) {
      __hip_atomic_fetch_add(ctr, 1u, __ATOMIC_ACQ_REL,
                             __HIP_MEMORY_SCOPE_AGENT);
      const unsigned tgt = (unsigned)(NWG * (s + 1));
      while (__hip_atomic_load(ctr, __ATOMIC_ACQUIRE,
                               __HIP_MEMORY_SCOPE_AGENT) < tgt) {
        __builtin_amdgcn_s_sleep(2);
      }
    }
    __syncthreads();
  }
}

extern "C" void kernel_launch(void* const* d_in, const int* in_sizes, int n_in,
                              void* d_out, int out_size, void* d_ws,
                              size_t ws_size, hipStream_t stream) {
  const float* seq  = (const float*)d_in[0];
  const float* w_ih = (const float*)d_in[1];
  const float* w_hh = (const float*)d_in[2];
  const float* b_ih = (const float*)d_in[3];
  const float* b_hh = (const float*)d_in[4];

  float* hs = (float*)d_out;
  float* cs = hs + (size_t)S * B * H;

  // Workspace: seq in f16 (33.5MB) + double-buffered h (128KB) + barrier ctr.
  _Float16* seq16 = (_Float16*)d_ws;
  _Float16* hbuf  = seq16 + (size_t)S * B * F;
  unsigned* ctr   = (unsigned*)(hbuf + (size_t)2 * B * H);

  lstm_init<<<512, 256, 0, stream>>>(seq, seq16, hbuf, ctr);

  const size_t shmem = (size_t)(2 * COLS * LDW) * sizeof(_Float16); // 260KB
  (void)hipFuncSetAttribute((const void*)lstm_recurrent,
                            hipFuncAttributeMaxDynamicSharedMemorySize,
                            (int)shmem);
  lstm_recurrent<<<NWG, 256, shmem, stream>>>(seq16, w_ih, w_hh, b_ih, b_hh,
                                              hs, cs, hbuf, ctr);
}
